// Attention_61538291417290
// MI455X (gfx1250) — matrix operand order repaired and checked by reading, not compiled, not run on hardware
//
#include <hip/hip_runtime.h>
#include <hip/hip_bf16.h>
#include <math.h>

typedef _Float16 v16h __attribute__((ext_vector_type(16)));
typedef _Float16 v8h  __attribute__((ext_vector_type(8)));
typedef _Float16 h4   __attribute__((ext_vector_type(4)));
typedef float    v8f  __attribute__((ext_vector_type(8)));

#define BM 128
#define BN 128
#define BK 32
#define LDS_STRIDE (BK + 8)   // 40 halves/row: 80B stride skews banks, keeps 16B align

// C[M,N] = alpha * (A[M,K_total] x B) + bias[n]
// BTRANS=true : B stored row-major [N,K] (ldb = K-dim row stride)  -> "A @ B^T"
// BTRANS=false: B stored row-major [K,N] (ldb = N-dim row stride)  -> "A @ B"
// Split-K: gridDim.z chunks; K = chunk length; chunk z writes C + z*partStride.
template <bool BTRANS>
__global__ __launch_bounds__(256)
void gemm_wmma_f16(const float* __restrict__ A, const float* __restrict__ B,
                   float* __restrict__ C, const float* __restrict__ bias,
                   int K, int lda, int ldb, int ldc, float alpha,
                   size_t partStride)
{
    __shared__ __align__(16) _Float16 sA[BM * LDS_STRIDE];
    __shared__ __align__(16) _Float16 sB[BN * LDS_STRIDE];

    const int tid   = threadIdx.x;
    const int lane  = tid & 31;
    const int wave  = tid >> 5;     // 0..7
    const int wrow  = wave >> 2;    // 0..1  -> 64-row slab
    const int wcol  = wave & 3;     // 0..3  -> 32-col slab
    const int lmod  = lane & 15;
    const int lhalf = lane >> 4;

    const int m0 = blockIdx.y * BM;
    const int n0 = blockIdx.x * BN;
    const int kb = blockIdx.z * K;            // split-K base
    C += (size_t)blockIdx.z * partStride;

    v8f acc[4][2] = {};
    float4 pa[4], pb[4];

    auto loadA = [&](int k0) {
        #pragma unroll
        for (int i = 0; i < 4; ++i) {
            int linear = i * 256 + tid;
            int r   = linear >> 3;
            int vec = linear & 7;
            pa[i] = *reinterpret_cast<const float4*>(
                &A[(size_t)(m0 + r) * lda + k0 + vec * 4]);
        }
    };
    auto loadB = [&](int k0) {
        #pragma unroll
        for (int i = 0; i < 4; ++i) {
            int linear = i * 256 + tid;
            if (BTRANS) {
                int r   = linear >> 3;      // n within tile
                int vec = linear & 7;       // k float4 index
                pb[i] = *reinterpret_cast<const float4*>(
                    &B[(size_t)(n0 + r) * ldb + k0 + vec * 4]);
            } else {
                int k   = linear >> 5;      // 0..31
                int vec = linear & 31;      // n float4 index
                pb[i] = *reinterpret_cast<const float4*>(
                    &B[(size_t)(k0 + k) * ldb + n0 + vec * 4]);
            }
        }
    };
    auto storeAB = [&]() {
        #pragma unroll
        for (int i = 0; i < 4; ++i) {
            int linear = i * 256 + tid;
            {   // A -> sA[r][k]
                int r   = linear >> 3;
                int vec = linear & 7;
                h4 hv;
                hv[0] = (_Float16)pa[i].x; hv[1] = (_Float16)pa[i].y;
                hv[2] = (_Float16)pa[i].z; hv[3] = (_Float16)pa[i].w;
                *reinterpret_cast<h4*>(&sA[r * LDS_STRIDE + vec * 4]) = hv;
            }
            if (BTRANS) {                   // B[N,K] -> sB[n][k] direct
                int r   = linear >> 3;
                int vec = linear & 7;
                h4 hv;
                hv[0] = (_Float16)pb[i].x; hv[1] = (_Float16)pb[i].y;
                hv[2] = (_Float16)pb[i].z; hv[3] = (_Float16)pb[i].w;
                *reinterpret_cast<h4*>(&sB[r * LDS_STRIDE + vec * 4]) = hv;
            } else {                        // B[K,N] -> sB[n][k] transpose
                int k   = linear >> 5;
                int vec = linear & 31;
                sB[(vec * 4 + 0) * LDS_STRIDE + k] = (_Float16)pb[i].x;
                sB[(vec * 4 + 1) * LDS_STRIDE + k] = (_Float16)pb[i].y;
                sB[(vec * 4 + 2) * LDS_STRIDE + k] = (_Float16)pb[i].z;
                sB[(vec * 4 + 3) * LDS_STRIDE + k] = (_Float16)pb[i].w;
            }
        }
    };

    const int nStages = K / BK;
    loadA(kb); loadB(kb);
    storeAB();
    __syncthreads();

    for (int kt = 0; kt < nStages; ++kt) {
        const bool hasNext = (kt + 1) < nStages;
        if (hasNext) {                       // prefetch next stage into regs;
            loadA(kb + (kt + 1) * BK);       // global loads hide under WMMAs
            loadB(kb + (kt + 1) * BK);
        }

        // ---- fragments per ISA 7.12.2 16-bit layouts + WMMA ----
        // B 32x16: lane n=lmod, K(e) = (e&8)*2 + lhalf*8 + (e&7) -> two 8-half runs
        v16h bfrag[2];
        #pragma unroll
        for (int ni = 0; ni < 2; ++ni) {
            int n = wcol * 32 + ni * 16 + lmod;
            union { v16h v; v8h h[2]; } u;
            u.h[0] = *reinterpret_cast<const v8h*>(&sB[n * LDS_STRIDE + lhalf * 8]);
            u.h[1] = *reinterpret_cast<const v8h*>(&sB[n * LDS_STRIDE + 16 + lhalf * 8]);
            bfrag[ni] = u.v;
        }
        // A 16x32: lane m=lmod, K(e) = (e&8)*2 + lhalf*8 + (e&7) -> two 8-half runs
        #pragma unroll
        for (int mi = 0; mi < 4; ++mi) {
            int m = wrow * 64 + mi * 16 + lmod;
            union { v16h v; v8h h[2]; } u;
            u.h[0] = *reinterpret_cast<const v8h*>(&sA[m * LDS_STRIDE + lhalf * 8]);
            u.h[1] = *reinterpret_cast<const v8h*>(&sA[m * LDS_STRIDE + 16 + lhalf * 8]);
            v16h afrag = u.v;
            #pragma unroll
            for (int ni = 0; ni < 2; ++ni) {
                acc[mi][ni] = __builtin_amdgcn_wmma_f32_16x16x32_f16(
                    false, afrag, false, bfrag[ni], (short)0, acc[mi][ni],
                    false, false);
            }
        }

        if (hasNext) {
            __syncthreads();                 // all waves done reading this stage
            storeAB();                       // regs -> LDS
            __syncthreads();                 // LDS ready for next stage
        }
    }

    // ---- epilogue: C/D layout row = r + 8*lhalf, col = lmod ----
    #pragma unroll
    for (int mi = 0; mi < 4; ++mi) {
        int mbase = m0 + wrow * 64 + mi * 16 + lhalf * 8;
        #pragma unroll
        for (int ni = 0; ni < 2; ++ni) {
            int col = n0 + wcol * 32 + ni * 16 + lmod;
            float bv = bias ? bias[col] : 0.0f;
            #pragma unroll
            for (int r = 0; r < 8; ++r) {
                C[(size_t)(mbase + r) * ldc + col] = acc[mi][ni][r] * alpha + bv;
            }
        }
    }
}

// outp[i] = alpha * sum_s parts[s*stride + i] + bias[i % ldc]
__global__ __launch_bounds__(256)
void reduce_partials(const float* __restrict__ parts, float* __restrict__ outp,
                     int S, size_t stride, float alpha,
                     const float* __restrict__ bias, int ldc)
{
    size_t i = (size_t)blockIdx.x * 256 + threadIdx.x;
    float s = 0.0f;
    for (int p = 0; p < S; ++p) s += parts[(size_t)p * stride + i];
    s *= alpha;
    if (bias) s += bias[i % (size_t)ldc];
    outp[i] = s;
}

// One 256-thread block per row; row cached in LDS; standard max-shifted softmax.
// MASKED: where(mask>0, x, -9e15), then diag=1.0 (matches reference order).
template <bool MASKED>
__global__ __launch_bounds__(256)
void softmax_row(const float* __restrict__ logits, const int* __restrict__ mask,
                 float* __restrict__ outp, int n)
{
    __shared__ float row[4096];
    __shared__ float redmax[8];
    __shared__ float redsum[8];
    const int i    = blockIdx.x;
    const int tid  = threadIdx.x;
    const int lane = tid & 31;
    const int wave = tid >> 5;

    float lmax = -3.0e38f;
    for (int j = tid; j < n; j += 256) {
        float v = logits[(size_t)i * n + j];
        if (MASKED) {
            if (mask[(size_t)i * n + j] <= 0) v = -9.0e15f;
            if (j == i) v = 1.0f;
        }
        row[j] = v;
        lmax = fmaxf(lmax, v);
    }
    #pragma unroll
    for (int o = 16; o > 0; o >>= 1) lmax = fmaxf(lmax, __shfl_xor(lmax, o, 32));
    if (lane == 0) redmax[wave] = lmax;
    __syncthreads();
    float rmax = redmax[0];
    #pragma unroll
    for (int w = 1; w < 8; ++w) rmax = fmaxf(rmax, redmax[w]);

    float lsum = 0.0f;
    for (int j = tid; j < n; j += 256) {
        float e = expf(row[j] - rmax);
        row[j] = e;
        lsum += e;
    }
    #pragma unroll
    for (int o = 16; o > 0; o >>= 1) lsum += __shfl_xor(lsum, o, 32);
    if (lane == 0) redsum[wave] = lsum;
    __syncthreads();
    float rsum = 0.0f;
    #pragma unroll
    for (int w = 0; w < 8; ++w) rsum += redsum[w];
    const float inv = 1.0f / rsum;
    for (int j = tid; j < n; j += 256) {
        outp[(size_t)i * n + j] = row[j] * inv;
    }
}

extern "C" void kernel_launch(void* const* d_in, const int* in_sizes, int n_in,
                              void* d_out, int out_size, void* d_ws, size_t ws_size,
                              hipStream_t stream)
{
    (void)in_sizes; (void)n_in; (void)out_size; (void)ws_size;

    const float* x     = (const float*)d_in[0];   // [4096, 512]
    const int*   mask  = (const int*)d_in[1];     // [4096, 4096]
    const float* w_qkv = (const float*)d_in[2];   // [768, 512]
    const float* w_out = (const float*)d_in[3];   // [512, 256]
    const float* b_out = (const float*)d_in[4];   // [512]

    const int N = 4096, DIM = 512, INNER = 256, QKV = 3 * INNER; // 768

    float* out  = (float*)d_out;                  // [4096, 512]
    float* attn = out + (size_t)N * DIM;          // [4096, 4096]
    float* dsm  = attn + (size_t)N * N;           // scaled logits -> softmax in place

    float* qkv   = (float*)d_ws;                  // [4096, 768]   12.6 MB
    float* oflat = qkv + (size_t)N * QKV;         // [4096, 256]    4.0 MB
    float* parts = oflat + (size_t)N * INNER;     // 4M floats     16.0 MB

    const float scale = 0.04419417382415922f;     // 512^-0.5
    dim3 blk(256);

    // 1) qkv = x @ w_qkv^T                      [4096,768]  (192 WGs)
    gemm_wmma_f16<true><<<dim3(QKV / BN, N / BM, 1), blk, 0, stream>>>(
        x, w_qkv, qkv, nullptr, DIM, DIM, DIM, QKV, 1.0f, 0);

    // 2) logits(head0) = scale * q0 @ k0^T      [4096,4096] (1024 WGs)
    //    q0 = qkv cols [0,128), k0 = qkv cols [256,384)
    gemm_wmma_f16<true><<<dim3(N / BN, N / BM, 1), blk, 0, stream>>>(
        qkv, qkv + 256, dsm, nullptr, 128, QKV, QKV, N, scale, 0);

    // 3) attn = softmax(where(mask>0, logits, -9e15) with diag=1)
    softmax_row<true><<<N, blk, 0, stream>>>(dsm, mask, attn, N);

    // 4) dots_softmax = softmax(logits)   (in place; after attn consumed logits)
    softmax_row<false><<<N, blk, 0, stream>>>(dsm, nullptr, dsm, N);

    // 5) oflat = attn @ v_flat   v_flat = qkv cols [512,768)
    //    split-K x4: 256 WGs, partials then deterministic reduction
    {
        const size_t pstride = (size_t)N * INNER;
        gemm_wmma_f16<false><<<dim3(INNER / BN, N / BM, 4), blk, 0, stream>>>(
            attn, qkv + 512, parts, nullptr, N / 4, N, QKV, INNER, 1.0f, pstride);
        reduce_partials<<<(N * INNER) / 256, blk, 0, stream>>>(
            parts, oflat, 4, pstride, 1.0f, nullptr, INNER);
    }

    // 6) out = oflat @ w_out^T + b_out   split-K x2: 256 WGs, bias in reduction
    {
        const size_t pstride = (size_t)N * DIM;
        gemm_wmma_f16<true><<<dim3(DIM / BN, N / BM, 2), blk, 0, stream>>>(
            oflat, w_out, parts, nullptr, INNER / 2, INNER, INNER, DIM, 1.0f, pstride);
        reduce_partials<<<(N * DIM) / 256, blk, 0, stream>>>(
            parts, out, 2, pstride, 1.0f, b_out, DIM);
    }
}
